// ElasticSolver_3401614098660
// MI455X (gfx1250) — matrix operand order repaired and checked by reading, not compiled, not run on hardware
//
#include <hip/hip_runtime.h>

#define BATCH 16384
#define DEFF  512

typedef _Float16 h16 __attribute__((ext_vector_type(16)));
typedef _Float16 h8  __attribute__((ext_vector_type(8)));
typedef float    f8  __attribute__((ext_vector_type(8)));

union H16U { h16 v; h8 h[2]; };

struct Segs   { const _Float16* p[5]; int str[5]; int len[5]; int n; };
struct Ranges { int start[5]; int len[5]; int n; };

__device__ __forceinline__ float gelu_f(float x) {
    return 0.5f * x * (1.0f + erff(x * 0.70710678118654752f));
}

__device__ __forceinline__ h16 load_a_frag(const _Float16* rowPtr, int kOff) {
    H16U u;
    u.h[0] = *(const h8*)(rowPtr + kOff);        // K = kOff .. kOff+7
    u.h[1] = *(const h8*)(rowPtr + kOff + 16);   // K = kOff+16 .. kOff+23
    return u.v;
}

// ---------------------------------------------------------------------------
// Segmented WMMA GEMM: out[M x N] = concat(A segments)[M x Ktot] @ W^T
// W is (N x Ktot) f16 row-major. Block = 256 thr (8 waves), each wave a 16x64
// output tile; grid = (M/128, N/64). Broadcast A rows via str==0.
// ---------------------------------------------------------------------------
__global__ __launch_bounds__(256) void es_wmma_gemm(Segs sg,
        const _Float16* __restrict__ w, int Ktot, int N,
        float* __restrict__ out) {
    const int lane = threadIdx.x & 31;
    const int wv   = threadIdx.x >> 5;
    const int l    = lane & 15;
    const int hi   = lane >> 4;
    const size_t mBase = (size_t)blockIdx.x * 128 + (size_t)wv * 16;
    const int    nBase = blockIdx.y * 64;

    f8 a0 = {}, a1 = {}, a2 = {}, a3 = {};
    int kglob = 0;
    for (int s = 0; s < sg.n; ++s) {
        const int st = sg.str[s];
        const _Float16* rowPtr = sg.p[s] + (st ? (mBase + (size_t)l) * (size_t)st : 0);
        const int L = sg.len[s];
        for (int kk = 0; kk < L; kk += 32) {
            h16 a = load_a_frag(rowPtr, kk + hi * 8);
            const _Float16* wp = w + (size_t)(nBase + l) * (size_t)Ktot
                                   + (size_t)(kglob + kk) + (size_t)hi * 8;
            h16 b0 = load_a_frag(wp, 0);
            h16 b1 = load_a_frag(wp + (size_t)16 * Ktot, 0);
            h16 b2 = load_a_frag(wp + (size_t)32 * Ktot, 0);
            h16 b3 = load_a_frag(wp + (size_t)48 * Ktot, 0);
            __builtin_prefetch(wp + 64, 0, 1);
            a0 = __builtin_amdgcn_wmma_f32_16x16x32_f16(false, a, false, b0, (short)0, a0, false, false);
            a1 = __builtin_amdgcn_wmma_f32_16x16x32_f16(false, a, false, b1, (short)0, a1, false, false);
            a2 = __builtin_amdgcn_wmma_f32_16x16x32_f16(false, a, false, b2, (short)0, a2, false, false);
            a3 = __builtin_amdgcn_wmma_f32_16x16x32_f16(false, a, false, b3, (short)0, a3, false, false);
        }
        kglob += L;
    }
    // C layout: VGPR r -> row (hi*8 + r), col = nBase + l (+ j*16 per acc)
    float* op = out + (mBase + (size_t)hi * 8) * (size_t)N + (size_t)(nBase + l);
#pragma unroll
    for (int r = 0; r < 8; ++r) {
        op[(size_t)r * N +  0] = a0[r];
        op[(size_t)r * N + 16] = a1[r];
        op[(size_t)r * N + 32] = a2[r];
        op[(size_t)r * N + 48] = a3[r];
    }
}

// ---------------------------------------------------------------------------
// Weight gather + f32->f16 convert (implements ROUTER/LATENT/EXEC_COLS)
// ---------------------------------------------------------------------------
__global__ void es_gather_cvt(const float* __restrict__ src, int srcRow,
                              _Float16* __restrict__ dst, int rows, int Ktot,
                              Ranges rg) {
    size_t idx = (size_t)blockIdx.x * blockDim.x + threadIdx.x;
    if (idx >= (size_t)rows * Ktot) return;
    int row = (int)(idx / Ktot);
    int k   = (int)(idx % Ktot);
    int off = 0, srcK = 0;
    for (int i = 0; i < rg.n; ++i) {
        if (k < off + rg.len[i]) { srcK = rg.start[i] + (k - off); break; }
        off += rg.len[i];
    }
    dst[idx] = (_Float16)src[(size_t)row * srcRow + srcK];
}

__global__ void es_op_gather(const float* __restrict__ op_emb,
                             const int* __restrict__ op_idx,
                             _Float16* __restrict__ opE) {
    size_t idx = (size_t)blockIdx.x * blockDim.x + threadIdx.x;   // B*512
    size_t row = idx >> 9;
    int    j   = (int)(idx & 511);
    opE[idx] = (_Float16)op_emb[(size_t)op_idx[row] * 1024 + j];
}

__global__ void es_copy_f32(const float* __restrict__ s, float* __restrict__ d, size_t n) {
    size_t i = (size_t)blockIdx.x * blockDim.x + threadIdx.x;
    if (i < n) d[i] = s[i];
}

// step_enc (dim 128, pos=step_num) + 4 rec_enc vectors (dim 64, pos=r)
__global__ void es_vec_init(const int* __restrict__ stepp,
                            _Float16* __restrict__ stepV, _Float16* __restrict__ recV) {
    int t = threadIdx.x;
    const float LN1E4 = 9.210340371976184f;
    if (t < 128) {
        int j = t >> 1;
        float div = expf(-(float)(2 * j) * LN1E4 / 128.0f);
        float pos = (float)stepp[0];
        stepV[t] = (_Float16)((t & 1) ? cosf(pos * div) : sinf(pos * div));
    }
    {   // 256 threads cover 4*64
        int rr = t >> 6, i = t & 63, j = i >> 1;
        float div = expf(-(float)(2 * j) * LN1E4 / 64.0f);
        float pos = (float)rr;
        recV[t] = (_Float16)((i & 1) ? cosf(pos * div) : sinf(pos * div));
    }
}

// ---------------------------------------------------------------------------
// wave-per-row LayerNorm(+bias) + GELU over 512 cols -> f16
// ---------------------------------------------------------------------------
__device__ __forceinline__ void ln_gelu_row(float v[16], int lane,
        const float* __restrict__ g, const float* __restrict__ be,
        _Float16* __restrict__ outRow) {
    float s = 0.f;
#pragma unroll
    for (int i = 0; i < 16; ++i) s += v[i];
#pragma unroll
    for (int o = 16; o > 0; o >>= 1) s += __shfl_xor(s, o, 32);
    float mean = s * (1.0f / 512.0f);
    float q = 0.f;
#pragma unroll
    for (int i = 0; i < 16; ++i) { float d = v[i] - mean; q += d * d; }
#pragma unroll
    for (int o = 16; o > 0; o >>= 1) q += __shfl_xor(q, o, 32);
    float rstd = rsqrtf(q * (1.0f / 512.0f) + 1e-5f);
#pragma unroll
    for (int i = 0; i < 16; ++i) {
        int c = lane + i * 32;
        float x = (v[i] - mean) * rstd * g[c] + be[c];
        outRow[c] = (_Float16)gelu_f(x);
    }
}

__global__ __launch_bounds__(256) void es_ln_gelu(const float* __restrict__ in,
        const float* __restrict__ bias, const float* __restrict__ g,
        const float* __restrict__ be, _Float16* __restrict__ out) {
    const int lane = threadIdx.x & 31;
    const size_t row = (size_t)blockIdx.x * 8 + (threadIdx.x >> 5);
    const float* rp = in + row * 512;
    float v[16];
#pragma unroll
    for (int i = 0; i < 16; ++i) { int c = lane + i * 32; v[i] = rp[c] + bias[c]; }
    ln_gelu_row(v, lane, g, be, out + row * 512);
}

// enc_state: e = W_se[:512] @ s + b_se -> LN -> GELU -> f16
__global__ __launch_bounds__(256) void es_state_enc(const float* __restrict__ st,
        const float* __restrict__ Wse, const float* __restrict__ bse,
        const float* __restrict__ g, const float* __restrict__ be,
        _Float16* __restrict__ out) {
    const int lane = threadIdx.x & 31;
    const size_t row = (size_t)blockIdx.x * 8 + (threadIdx.x >> 5);
    float s0 = st[row * 3 + 0], s1 = st[row * 3 + 1], s2 = st[row * 3 + 2];
    float v[16];
#pragma unroll
    for (int i = 0; i < 16; ++i) {
        int n = lane + i * 32;
        v[i] = Wse[n * 3] * s0 + Wse[n * 3 + 1] * s1 + Wse[n * 3 + 2] * s2 + bse[n];
    }
    ln_gelu_row(v, lane, g, be, out + row * 512);
}

__global__ void es_bias_act(const float* __restrict__ in, const float* __restrict__ bias,
                            _Float16* __restrict__ out, int N, int doGelu) {
    size_t idx = (size_t)blockIdx.x * blockDim.x + threadIdx.x;   // B*N exactly
    int col = (int)(idx % (size_t)N);
    float x = in[idx] + bias[col];
    if (doGelu) x = gelu_f(x);
    out[idx] = (_Float16)x;
}

// logits = r1 @ W_r2[:, :128]^T + b_r2 ; route = argmax (first on ties)
__global__ void es_route(const _Float16* __restrict__ r1, const float* __restrict__ Wr2,
                         const float* __restrict__ br2, int* __restrict__ route) {
    size_t row = (size_t)blockIdx.x * blockDim.x + threadIdx.x;
    const _Float16* rp = r1 + row * 128;
    float best = -3.4e38f; int bi = 0;
    for (int j = 0; j < 4; ++j) {
        float acc = br2[j];
        const float* wr = Wr2 + j * 256;
        for (int k = 0; k < 128; ++k) acc += (float)rp[k] * wr[k];
        if (acc > best) { best = acc; bi = j; }
    }
    route[row] = bi;
}

// comp features + comp_enc (N=128, K=6)
__global__ void es_comp(const float* __restrict__ st, const float* __restrict__ Wc,
                        const float* __restrict__ bc, _Float16* __restrict__ out) {
    size_t idx = (size_t)blockIdx.x * blockDim.x + threadIdx.x;   // B*128
    size_t row = idx >> 7;
    int n = (int)(idx & 127);
    float x0 = st[row * 3], x1 = st[row * 3 + 1], x2 = st[row * 3 + 2];
    float c[6] = { x0 > x1 ? 1.f : 0.f, x1 > x2 ? 1.f : 0.f, x2 > x0 ? 1.f : 0.f,
                   (x0 - x1) * (1.f / 30.f), (x1 - x2) * (1.f / 30.f), (x2 - x0) * (1.f / 30.f) };
    float acc = bc[n];
    const float* wr = Wc + n * 6;
#pragma unroll
    for (int i = 0; i < 6; ++i) acc += c[i] * wr[i];
    out[idx] = (_Float16)gelu_f(acc);
}

// delta = e2 @ W_e3[:, :512]^T + b_e3 ; rec += delta ; emit on matching route
__global__ void es_update(const _Float16* __restrict__ e2, const float* __restrict__ We3,
                          const float* __restrict__ be3, const int* __restrict__ route,
                          int r, float* __restrict__ recSt, float* __restrict__ outp) {
    size_t row = (size_t)blockIdx.x * blockDim.x + threadIdx.x;
    const _Float16* ep = e2 + row * 512;
    float d[3];
#pragma unroll
    for (int v = 0; v < 3; ++v) {
        float acc = be3[v];
        const float* wr = We3 + (size_t)v * 1024;
        for (int k = 0; k < 512; ++k) acc += (float)ep[k] * wr[k];
        d[v] = acc;
    }
    float n0 = recSt[row * 3 + 0] + d[0];
    float n1 = recSt[row * 3 + 1] + d[1];
    float n2 = recSt[row * 3 + 2] + d[2];
    recSt[row * 3 + 0] = n0; recSt[row * 3 + 1] = n1; recSt[row * 3 + 2] = n2;
    if (route[row] == r) { outp[row * 3] = n0; outp[row * 3 + 1] = n1; outp[row * 3 + 2] = n2; }
}

// ---------------------------------------------------------------------------
extern "C" void kernel_launch(void* const* d_in, const int* in_sizes, int n_in,
                              void* d_out, int out_size, void* d_ws, size_t ws_size,
                              hipStream_t stream) {
    (void)in_sizes; (void)n_in; (void)out_size; (void)ws_size;
    const float* state  = (const float*)d_in[0];
    const float* wisdom = (const float*)d_in[1];
    const int*   op_idx = (const int*)d_in[2];
    const int*   step_n = (const int*)d_in[3];
    const float* op_emb = (const float*)d_in[4];
    const float* W_se = (const float*)d_in[5],  *b_se = (const float*)d_in[6];
    const float* g_se = (const float*)d_in[7],  *be_se= (const float*)d_in[8];
    const float* W_wis= (const float*)d_in[9],  *b_wis= (const float*)d_in[10];
    const float* W_r1 = (const float*)d_in[11], *b_r1 = (const float*)d_in[12];
    const float* W_r2 = (const float*)d_in[13], *b_r2 = (const float*)d_in[14];
    const float* W_lg1= (const float*)d_in[15], *b_lg1= (const float*)d_in[16];
    const float* g_lg = (const float*)d_in[17], *be_lg= (const float*)d_in[18];
    const float* W_lg2= (const float*)d_in[19], *b_lg2= (const float*)d_in[20];
    const float* W_cmp= (const float*)d_in[21], *b_cmp= (const float*)d_in[22];
    const float* W_le = (const float*)d_in[23], *b_le = (const float*)d_in[24];
    const float* g_le = (const float*)d_in[25], *be_le= (const float*)d_in[26];
    const float* W_e1 = (const float*)d_in[27], *b_e1 = (const float*)d_in[28];
    const float* g_ex = (const float*)d_in[29], *be_ex= (const float*)d_in[30];
    const float* W_e2 = (const float*)d_in[31], *b_e2 = (const float*)d_in[32];
    const float* W_e3 = (const float*)d_in[33], *b_e3 = (const float*)d_in[34];
    float* outp = (float*)d_out;

    char* base = (char*)d_ws; size_t off = 0;
    auto alloc = [&](size_t bytes) -> void* {
        void* p = base + off; off += (bytes + 255) & ~(size_t)255; return p;
    };
    const size_t B = BATCH;
    _Float16* wWis = (_Float16*)alloc(256 * 256 * 2);
    _Float16* wR1  = (_Float16*)alloc(128 * 1024 * 2);
    _Float16* wLg1 = (_Float16*)alloc(512 * 1472 * 2);
    _Float16* wLg2 = (_Float16*)alloc(256 * 512 * 2);
    _Float16* wLe  = (_Float16*)alloc(512 * 256 * 2);
    _Float16* wE1  = (_Float16*)alloc(512 * 1152 * 2);
    _Float16* wE2  = (_Float16*)alloc(512 * 512 * 2);
    _Float16* stepV= (_Float16*)alloc(128 * 2);
    _Float16* recV = (_Float16*)alloc(256 * 2);
    _Float16* opE  = (_Float16*)alloc(B * 512 * 2);
    _Float16* wisH = (_Float16*)alloc(B * 256 * 2);
    _Float16* wisE = (_Float16*)alloc(B * 256 * 2);
    _Float16* seB  = (_Float16*)alloc(B * 512 * 2);
    _Float16* t0   = (_Float16*)alloc(B * 512 * 2);
    _Float16* t1   = (_Float16*)alloc(B * 256 * 2);
    _Float16* cmpB = (_Float16*)alloc(B * 128 * 2);   // also r1
    _Float16* e1B  = (_Float16*)alloc(B * 512 * 2);
    float*    f32o = (float*)alloc(B * 512 * 4);
    float*    recSt= (float*)alloc(B * 3 * 4);
    int*      route= (int*)alloc(B * 4);

    auto cvt = [&](const float* src, int srcRow, _Float16* dst, int rows, int Ktot, Ranges rg) {
        size_t tot = (size_t)rows * Ktot;
        es_gather_cvt<<<(unsigned)((tot + 255) / 256), 256, 0, stream>>>(src, srcRow, dst, rows, Ktot, rg);
    };
    cvt(W_wis, 256,  wWis, 256, 256,  Ranges{{0},{256},1});
    cvt(W_r1,  2048, wR1,  128, 1024, Ranges{{0,1024},{512,512},2});
    cvt(W_lg1, 2880, wLg1, 512, 1472, Ranges{{0,1024,2048,2304,2816},{512,512,128,256,64},5});
    cvt(W_lg2, 1024, wLg2, 256, 512,  Ranges{{0},{512},1});
    cvt(W_le,  512,  wLe,  512, 256,  Ranges{{0},{256},1});
    cvt(W_e1,  2176, wE1,  512, 1152, Ranges{{0,1024,2048},{512,512,128},3});
    cvt(W_e2,  1024, wE2,  512, 512,  Ranges{{0},{512},1});
    cvt(wisdom,256,  wisH, (int)B, 256, Ranges{{0},{256},1});

    es_vec_init<<<1, 256, 0, stream>>>(step_n, stepV, recV);
    es_op_gather<<<(unsigned)(B * 512 / 256), 256, 0, stream>>>(op_emb, op_idx, opE);
    es_copy_f32<<<(unsigned)((B * 3 + 255) / 256), 256, 0, stream>>>(state, recSt, B * 3);

    auto gemm = [&](Segs sg, const _Float16* w, int Ktot, int N, float* out) {
        dim3 grid((unsigned)(B / 128), (unsigned)(N / 64));
        es_wmma_gemm<<<grid, 256, 0, stream>>>(sg, w, Ktot, N, out);
    };
    auto bact = [&](const float* in, const float* bias, _Float16* out, int N, int doGelu) {
        es_bias_act<<<(unsigned)(B * (size_t)N / 256), 256, 0, stream>>>(in, bias, out, N, doGelu);
    };
    auto lng = [&](const float* in, const float* bias, const float* g, const float* be, _Float16* out) {
        es_ln_gelu<<<(unsigned)(B / 8), 256, 0, stream>>>(in, bias, g, be, out);
    };

    // wisdom_enc = gelu(wisdom @ W_wis[:256]^T + b_wis)
    gemm(Segs{{wisH},{256},{256},1}, wWis, 256, 256, f32o);
    bact(f32o, b_wis, wisE, 256, 1);

    // router: se(initial), r1, logits/argmax
    es_state_enc<<<(unsigned)(B / 8), 256, 0, stream>>>(recSt, W_se, b_se, g_se, be_se, seB);
    gemm(Segs{{seB, opE},{512,512},{512,512},2}, wR1, 1024, 128, f32o);
    bact(f32o, b_r1, cmpB, 128, 1);                    // cmpB temporarily = r1
    es_route<<<(unsigned)(B / 256), 256, 0, stream>>>(cmpB, W_r2, b_r2, route);

    for (int r = 0; r < 4; ++r) {
        es_state_enc<<<(unsigned)(B / 8), 256, 0, stream>>>(recSt, W_se, b_se, g_se, be_se, seB);
        es_comp<<<(unsigned)(B * 128 / 256), 256, 0, stream>>>(recSt, W_cmp, b_cmp, cmpB);
        // lg1 = gelu(ln(li @ Wlg1^T + b_lg1)); li = [se, opE, step, wisE, rec]
        gemm(Segs{{seB, opE, stepV, wisE, recV + r * 64},
                  {512, 512, 0, 256, 0},
                  {512, 512, 128, 256, 64}, 5}, wLg1, 1472, 512, f32o);
        lng(f32o, b_lg1, g_lg, be_lg, t0);
        // latent = lg1 @ W_lg2[:256,:512]^T + b_lg2
        gemm(Segs{{t0},{512},{512},1}, wLg2, 512, 256, f32o);
        bact(f32o, b_lg2, t1, 256, 0);
        // le = gelu(ln(latent @ W_le[:512,:256]^T + b_le))
        gemm(Segs{{t1},{256},{256},1}, wLe, 256, 512, f32o);
        lng(f32o, b_le, g_le, be_le, t0);
        // e1 = gelu(ln(ei @ We1^T + b_e1)); ei = [se, le, comp_enc]
        gemm(Segs{{seB, t0, cmpB},{512,512,128},{512,512,128},3}, wE1, 1152, 512, f32o);
        lng(f32o, b_e1, g_ex, be_ex, e1B);
        // e2 = gelu(e1 @ W_e2[:512,:512]^T + b_e2)
        gemm(Segs{{e1B},{512},{512},1}, wE2, 512, 512, f32o);
        bact(f32o, b_e2, t0, 512, 1);
        // delta + state update + routed output
        es_update<<<(unsigned)(B / 256), 256, 0, stream>>>(t0, W_e3, b_e3, route, r, recSt, outp);
    }
}
